// CNN_RNN_34574486733030
// MI455X (gfx1250) — compile-verified, matrix operand order checked by reading, not run on hardware
//
#include <hip/hip_runtime.h>
#include <hip/hip_bf16.h>
#include <math.h>

// ---------------------------------------------------------------------------
// Problem constants (match the reference)
// ---------------------------------------------------------------------------
#define NFRAMES   2048
#define HGRU      1920
#define HGRU3     5760       // 3*HGRU
#define FEAT      1920       // 128*3*5
// conv1: [1,28,46] -> conv 9x9 pad2 -> [64,24,42] -> sig -> pool3 -> [64,8,14]
// conv2: [64,8,14] -> conv 4x4 pad2 -> [128,9,15] -> sig -> pool3 -> [128,3,5]
#define C1OUT_PER_IMG (64*8*14)      // 7168
#define NWG_GRU   240
#define ROWS_PER_WG (HGRU3 / NWG_GRU)  // 24

typedef unsigned int u32;
typedef unsigned short u16;
typedef __attribute__((ext_vector_type(16))) __bf16 v16bf;
typedef __attribute__((ext_vector_type(8)))  float  v8f;
typedef __attribute__((ext_vector_type(4)))  u32    u32x4;
typedef __attribute__((ext_vector_type(8)))  int    i32x8;
typedef __attribute__((ext_vector_type(4)))  int    i32x4;

// ---------------------------------------------------------------------------
// small helpers
// ---------------------------------------------------------------------------
__device__ __forceinline__ float sigf(float x) { return 1.0f / (1.0f + __expf(-x)); }

__device__ __forceinline__ u16 f2bf(float f) {
    u32 u = __builtin_bit_cast(u32, f);
    u32 r = u + 0x7fffu + ((u >> 16) & 1u);     // round-to-nearest-even
    return (u16)(r >> 16);
}
__device__ __forceinline__ float bf2f(u16 b) {
    u32 u = ((u32)b) << 16;
    return __builtin_bit_cast(float, u);
}

// wave32 all-lanes sum via ds_swizzle (xor butterflies, and_mask=0x1f)
#define SWZ_ADD(v, imm)                                                          \
    v += __builtin_bit_cast(float,                                               \
            __builtin_amdgcn_ds_swizzle(__builtin_bit_cast(int, v), imm))
__device__ __forceinline__ float wave_sum(float v) {
    SWZ_ADD(v, 0x401f);  // xor 16
    SWZ_ADD(v, 0x201f);  // xor 8
    SWZ_ADD(v, 0x101f);  // xor 4
    SWZ_ADD(v, 0x081f);  // xor 2
    SWZ_ADD(v, 0x041f);  // xor 1
    return v;
}

// ---------------------------------------------------------------------------
// TDM: async 2D tile load (global -> LDS).  D# packed per CDNA5 ISA §8.
// data_size = 2 bytes (bf16).  tile = tile_k elems wide x tile_rows rows,
// row stride = row_stride elems.  lds_off = byte offset in this WG's LDS.
// ---------------------------------------------------------------------------
__device__ __forceinline__ void tdm_load_2d_bf16(u32 lds_off, const void* gptr,
                                                 u32 tile_k, u32 tile_rows,
                                                 u32 row_stride) {
    unsigned long long ga = (unsigned long long)gptr;
    u32x4 g0;
    g0.x = 1u;                                           // count=1 (valid user D#)
    g0.y = lds_off;                                      // lds_addr
    g0.z = (u32)ga;                                      // global_addr[31:0]
    g0.w = ((u32)(ga >> 32) & 0x01FFFFFFu) | (2u << 30); // addr[56:32] | type=2
    u32 dim0 = row_stride;                               // tensor_dim0 (K extent)
    u32 dim1 = tile_rows;                                // tensor_dim1
    unsigned long long s0 = (unsigned long long)row_stride;
    i32x8 g1;
    g1[0] = (int)(1u << 16);                                               // data_size=2B
    g1[1] = (int)((dim0 & 0xFFFFu) << 16);                                 // dim0[15:0]
    g1[2] = (int)(((dim0 >> 16) & 0xFFFFu) | ((dim1 & 0xFFFFu) << 16));    // dim0 hi | dim1 lo
    g1[3] = (int)(((dim1 >> 16) & 0xFFFFu) | ((tile_k & 0xFFFFu) << 16));  // dim1 hi | tile_dim0
    g1[4] = (int)(tile_rows & 0xFFFFu);                                    // tile_dim1 (tile_dim2=0)
    g1[5] = (int)(u32)(s0 & 0xFFFFFFFFull);                                // dim0_stride[31:0]
    g1[6] = (int)(u32)((s0 >> 32) & 0xFFFFull);                            // dim0_stride[47:32]
    g1[7] = 0;
    i32x4 z4 = {0, 0, 0, 0};
#if __has_include(<hip/amd_detail/amd_gfx1250_TDM.h>)
    i32x8 z8 = {0, 0, 0, 0, 0, 0, 0, 0};
    __builtin_amdgcn_tensor_load_to_lds(g0, g1, z4, z4, z8, 0);
#else
    __builtin_amdgcn_tensor_load_to_lds(g0, g1, z4, z4, 0);
#endif
}

// ---------------------------------------------------------------------------
// Kernel 0: fp32 -> bf16 weight casts (w_ih, w_hh)
// ---------------------------------------------------------------------------
__global__ void cast_weights_bf16(const float* __restrict__ wih,
                                  const float* __restrict__ whh,
                                  u16* __restrict__ wih_bf,
                                  u16* __restrict__ whh_bf, int n) {
    for (int i = blockIdx.x * blockDim.x + threadIdx.x; i < n;
         i += gridDim.x * blockDim.x) {
        wih_bf[i] = f2bf(wih[i]);
        whh_bf[i] = f2bf(whh[i]);
    }
}

// ---------------------------------------------------------------------------
// Kernel 1: conv1 (9x9 pad2) + sigmoid + maxpool3 — one image per block
// out layout: [n][c][py][px] = [2048][64][8][14] fp32
// ---------------------------------------------------------------------------
__global__ void conv1_sig_pool(const float* __restrict__ x,
                               const float* __restrict__ w,
                               const float* __restrict__ b,
                               float* __restrict__ out) {
    __shared__ float simg[28 * 46];      // 1288
    __shared__ float swt[64 * 81];       // 5184
    const int n   = blockIdx.x;
    const int tid = threadIdx.x;
    const float* img = x + (size_t)n * 28 * 46;
    for (int i = tid; i < 28 * 46; i += blockDim.x) simg[i] = img[i];
    for (int i = tid; i < 64 * 81; i += blockDim.x) swt[i]  = w[i];
    __syncthreads();

    for (int idx = tid; idx < C1OUT_PER_IMG; idx += blockDim.x) {
        const int c   = idx / 112;
        const int rem = idx - c * 112;
        const int py  = rem / 14;
        const int px  = rem - py * 14;
        const float* wc = swt + c * 81;
        float mx = -3.4e38f;
        #pragma unroll
        for (int dy = 0; dy < 3; ++dy)
        #pragma unroll
        for (int dx = 0; dx < 3; ++dx) {
            const int y  = py * 3 + dy;          // 0..23
            const int xc = px * 3 + dx;          // 0..41
            float s = b[c];
            #pragma unroll
            for (int i = 0; i < 9; ++i) {
                const int iy = y - 2 + i;
                if (iy < 0 || iy >= 28) continue;
                #pragma unroll
                for (int j = 0; j < 9; ++j) {
                    const int ix = xc - 2 + j;
                    if (ix < 0 || ix >= 46) continue;
                    s = fmaf(simg[iy * 46 + ix], wc[i * 9 + j], s);
                }
            }
            const float v = sigf(s);
            mx = fmaxf(mx, v);
        }
        out[(size_t)n * C1OUT_PER_IMG + idx] = mx;
    }
}

// ---------------------------------------------------------------------------
// Kernel 2: conv2 (4x4 pad2 over 64ch) + sigmoid + maxpool3, store feats bf16
// feats layout (C-major flatten, matches torch .view): [n][c*15 + py*5 + px]
// ---------------------------------------------------------------------------
__global__ void conv2_sig_pool(const float* __restrict__ c1out,
                               const float* __restrict__ w2,
                               const float* __restrict__ b2,
                               u16* __restrict__ feats_bf) {
    __shared__ float sin[64 * 8 * 14];   // 7168 floats = 28 KB
    const int n   = blockIdx.x;
    const int tid = threadIdx.x;
    const float* src = c1out + (size_t)n * C1OUT_PER_IMG;
    for (int i = tid; i < C1OUT_PER_IMG; i += blockDim.x) sin[i] = src[i];
    __syncthreads();

    for (int idx = tid; idx < FEAT; idx += blockDim.x) {
        const int c   = idx / 15;
        const int rem = idx - c * 15;
        const int py  = rem / 5;
        const int px  = rem - py * 5;
        const float* wc = w2 + (size_t)c * 64 * 16;
        float mx = -3.4e38f;
        #pragma unroll
        for (int dy = 0; dy < 3; ++dy)
        #pragma unroll
        for (int dx = 0; dx < 3; ++dx) {
            const int y  = py * 3 + dy;          // 0..8
            const int xc = px * 3 + dx;          // 0..14
            float s = b2[c];
            for (int ic = 0; ic < 64; ++ic) {
                const float* lin = sin + ic * 112;
                const float* wi  = wc + ic * 16;
                #pragma unroll
                for (int i = 0; i < 4; ++i) {
                    const int iy = y - 2 + i;
                    if (iy < 0 || iy >= 8) continue;
                    #pragma unroll
                    for (int j = 0; j < 4; ++j) {
                        const int ix = xc - 2 + j;
                        if (ix < 0 || ix >= 14) continue;
                        s = fmaf(lin[iy * 14 + ix], wi[i * 4 + j], s);
                    }
                }
            }
            mx = fmaxf(mx, sigf(s));
        }
        feats_bf[(size_t)n * FEAT + idx] = f2bf(mx);
    }
}

// ---------------------------------------------------------------------------
// Kernel 3: x_proj GEMM.  C[2048,5760] = A[2048,1920]bf16 * B[5760,1920]^T bf16
// + b_ih.
// Block tile 128x128, 8 waves as 4x2, each wave 32x64 via 2x4 WMMA tiles.
// A/B tiles staged into LDS by the Tensor Data Mover, DOUBLE-BUFFERED so the
// DMA of tile kt+1 overlaps the WMMAs of tile kt.
// LDS layout per buffer: A 128x32 bf16 (8KB) then B 128x32 bf16 (8KB).
// ---------------------------------------------------------------------------
#define GEMM_BUF_BYTES 16384u          // one double-buffer half (A+B)
__global__ void xproj_gemm_wmma(const u16* __restrict__ feats_bf,
                                const u16* __restrict__ wih_bf,
                                const float* __restrict__ b_ih,
                                float* __restrict__ xproj) {
    extern __shared__ char smem[];                 // 2 * 16KB
    const int tid    = threadIdx.x;
    const int wid    = tid >> 5;
    const int lane   = tid & 31;
    const int blockM = blockIdx.y * 128;
    const int blockN = blockIdx.x * 128;
    const int m0     = (wid >> 1) * 32;            // wave tile origin in block tile
    const int n0     = (wid & 1) * 64;

    v8f acc[2][4] = {};
    const int mlane = lane & 15;
    const int kof   = (lane >> 4) * 8;             // 16-bit A/B lane K-chunk origin

    const u16* Ag = feats_bf + (size_t)blockM * HGRU;
    const u16* Bg = wih_bf  + (size_t)blockN * HGRU;

    // prologue: DMA tile 0 into buffer 0
    if (wid == 0) {
        tdm_load_2d_bf16(0u,    Ag, 32, 128, HGRU);
        tdm_load_2d_bf16(8192u, Bg, 32, 128, HGRU);
    }

    for (int kt = 0; kt < HGRU / 32; ++kt) {
        const u32 cur = (u32)(kt & 1) * GEMM_BUF_BYTES;
        const u32 nxt = (u32)((kt + 1) & 1) * GEMM_BUF_BYTES;
        if (wid == 0) __builtin_amdgcn_s_wait_tensorcnt(0);  // tile kt resident
        __syncthreads();
        // overlap: DMA tile kt+1 while computing tile kt
        if (wid == 0 && kt + 1 < HGRU / 32) {
            const int k = (kt + 1) * 32;
            tdm_load_2d_bf16(nxt,         Ag + k, 32, 128, HGRU);
            tdm_load_2d_bf16(nxt + 8192u, Bg + k, 32, 128, HGRU);
        }

        const __bf16* As = (const __bf16*)(smem + cur);
        const __bf16* Bs = (const __bf16*)(smem + cur + 8192u);

        // fragments per ISA 16-bit 16x32 layout:
        // lanes 0-15: K 0..7 & 16..23 ; lanes 16-31: K 8..15 & 24..31
        v16bf a[2], bb[4];
        #pragma unroll
        for (int s = 0; s < 2; ++s) {
            const __bf16* pa = As + (m0 + s * 16 + mlane) * 32 + kof;
            #pragma unroll
            for (int i = 0; i < 8; ++i) {
                a[s][i]     = pa[i];
                a[s][8 + i] = pa[16 + i];
            }
        }
        #pragma unroll
        for (int s = 0; s < 4; ++s) {
            const __bf16* pb = Bs + (n0 + s * 16 + mlane) * 32 + kof;
            #pragma unroll
            for (int i = 0; i < 8; ++i) {
                bb[s][i]     = pb[i];
                bb[s][8 + i] = pb[16 + i];
            }
        }
        #pragma unroll
        for (int ms = 0; ms < 2; ++ms)
        #pragma unroll
        for (int ns = 0; ns < 4; ++ns)
            acc[ms][ns] = __builtin_amdgcn_wmma_f32_16x16x32_bf16(
                false, a[ms], false, bb[ns], (short)0, acc[ms][ns], false, false);
        __syncthreads();   // all reads of buffer `cur` done before it is refilled
    }

    // C layout: VGPR j -> M = j + 8*(lane>=16), N = lane&15
    const int rlo  = (lane >> 4) * 8;
    const int ncol = lane & 15;
    #pragma unroll
    for (int ms = 0; ms < 2; ++ms)
    #pragma unroll
    for (int ns = 0; ns < 4; ++ns) {
        const int col   = blockN + n0 + ns * 16 + ncol;
        const float bia = b_ih[col];
        #pragma unroll
        for (int j = 0; j < 8; ++j) {
            const int row = blockM + m0 + ms * 16 + rlo + j;
            xproj[(size_t)row * HGRU3 + col] = acc[ms][ns][j] + bia;
        }
    }
}

// ---------------------------------------------------------------------------
// Kernel 4: init GRU state + device barrier counters
// ---------------------------------------------------------------------------
__global__ void gru_init(const float* __restrict__ h0, float* __restrict__ hbuf,
                         u32* __restrict__ bar) {
    const int i = blockIdx.x * blockDim.x + threadIdx.x;
    if (i < HGRU) hbuf[i] = h0[i];
    if (i < 4) bar[i] = 0;
}

// ---------------------------------------------------------------------------
// device-wide barrier (sense via generation counter)
// ---------------------------------------------------------------------------
__device__ __forceinline__ void grid_sync(u32* cnt, u32* gen, u32 nwg) {
    __threadfence();
    __syncthreads();
    if (threadIdx.x == 0) {
        const u32 g = __atomic_load_n(gen, __ATOMIC_ACQUIRE);
        if (atomicAdd(cnt, 1u) == nwg - 1u) {
            __atomic_store_n(cnt, 0u, __ATOMIC_RELAXED);
            __atomic_store_n(gen, g + 1u, __ATOMIC_RELEASE);
        } else {
            while (__atomic_load_n(gen, __ATOMIC_ACQUIRE) == g)
                __builtin_amdgcn_s_sleep(1);
        }
    }
    __syncthreads();
}

// ---------------------------------------------------------------------------
// Kernel 5: persistent GRU scan.  240 WGs x 256 thr.
// Phase 1: hp = w_hh(bf16) . h + b_hh  (wave per row, ds_swizzle reduce)
// Phase 2: gate math, write h_next.  Device barrier between phases/steps.
// ---------------------------------------------------------------------------
__global__ void gru_scan(const u16* __restrict__ whh_bf,
                         const float* __restrict__ b_hh,
                         const float* __restrict__ xproj,
                         float* __restrict__ hbuf,      // 2 x 1920
                         float* __restrict__ hp,        // 5760
                         u32* __restrict__ bar, int T) {
    __shared__ float hl[HGRU];
    const int tid  = threadIdx.x;
    const int wid  = tid >> 5;
    const int lane = tid & 31;
    u32* cnt = bar + 0;
    u32* gen = bar + 1;

    for (int t = 0; t < T; ++t) {
        const float* hold  = hbuf + (t & 1) * HGRU;
        float*       hnext = hbuf + ((t + 1) & 1) * HGRU;

        for (int i = tid; i < HGRU; i += blockDim.x) hl[i] = hold[i];
        __syncthreads();

        // phase 1: 24 rows per WG, 3 per wave
        #pragma unroll
        for (int rr = 0; rr < ROWS_PER_WG / 8; ++rr) {
            const int r = blockIdx.x * ROWS_PER_WG + wid * (ROWS_PER_WG / 8) + rr;
            const u16* wr = whh_bf + (size_t)r * HGRU;
            __builtin_prefetch(wr + HGRU, 0, 1);      // global_prefetch next row
            float acc = 0.0f;
            #pragma unroll
            for (int it = 0; it < HGRU / 128; ++it) { // 15 iters, 4 bf16/lane/iter
                const int k = (it * 32 + lane) * 4;
                const unsigned long long w4 =
                    *(const unsigned long long*)(wr + k);
                acc = fmaf(bf2f((u16)(w4 >>  0)), hl[k + 0], acc);
                acc = fmaf(bf2f((u16)(w4 >> 16)), hl[k + 1], acc);
                acc = fmaf(bf2f((u16)(w4 >> 32)), hl[k + 2], acc);
                acc = fmaf(bf2f((u16)(w4 >> 48)), hl[k + 3], acc);
            }
            acc = wave_sum(acc);
            if (lane == 0) hp[r] = acc + b_hh[r];
        }
        grid_sync(cnt, gen, NWG_GRU);

        // phase 2: 1920 lanes total = 8 per WG
        if (tid < HGRU / NWG_GRU) {
            const int i = blockIdx.x * (HGRU / NWG_GRU) + tid;
            const float* xp = xproj + (size_t)t * HGRU3;
            const float rg = sigf(xp[i]            + hp[i]);
            const float zg = sigf(xp[HGRU + i]     + hp[HGRU + i]);
            const float ng = tanhf(xp[2 * HGRU + i] + rg * hp[2 * HGRU + i]);
            hnext[i] = (1.0f - zg) * ng + zg * hold[i];
        }
        grid_sync(cnt, gen, NWG_GRU);
    }
}

// ---------------------------------------------------------------------------
// Kernel 6: FC head — 2 classes, one wave each
// ---------------------------------------------------------------------------
__global__ void fc_head(const float* __restrict__ h,
                        const float* __restrict__ fcw,
                        const float* __restrict__ fcb,
                        float* __restrict__ out) {
    const int wid  = threadIdx.x >> 5;
    const int lane = threadIdx.x & 31;
    if (wid >= 2) return;
    const float* wr = fcw + wid * HGRU;
    float acc = 0.0f;
    for (int k = lane; k < HGRU; k += 32) acc = fmaf(h[k], wr[k], acc);
    acc = wave_sum(acc);
    if (lane == 0) out[wid] = acc + fcb[wid];
}

// ---------------------------------------------------------------------------
// launch
// ---------------------------------------------------------------------------
extern "C" void kernel_launch(void* const* d_in, const int* in_sizes, int n_in,
                              void* d_out, int out_size, void* d_ws, size_t ws_size,
                              hipStream_t stream) {
    const float* x       = (const float*)d_in[0];
    const float* conv1_w = (const float*)d_in[1];
    const float* conv1_b = (const float*)d_in[2];
    const float* conv2_w = (const float*)d_in[3];
    const float* conv2_b = (const float*)d_in[4];
    const float* w_ih    = (const float*)d_in[5];
    const float* w_hh    = (const float*)d_in[6];
    const float* b_ih    = (const float*)d_in[7];
    const float* b_hh    = (const float*)d_in[8];
    const float* fc_w    = (const float*)d_in[9];
    const float* fc_b    = (const float*)d_in[10];
    const float* h0      = (const float*)d_in[11];
    float* out = (float*)d_out;

    // workspace carve (all chunks 256B-multiple)
    char* w = (char*)d_ws;
    float* conv1out = (float*)w;  w += (size_t)NFRAMES * C1OUT_PER_IMG * 4;   // 58,720,256
    u16* feats_bf   = (u16*)w;    w += (size_t)NFRAMES * FEAT * 2;            //  7,864,320
    u16* wih_bf     = (u16*)w;    w += (size_t)HGRU3 * HGRU * 2;              // 22,118,400
    u16* whh_bf     = (u16*)w;    w += (size_t)HGRU3 * HGRU * 2;              // 22,118,400
    float* xproj    = (float*)w;  w += (size_t)NFRAMES * HGRU3 * 4;           // 47,185,920
    float* hp       = (float*)w;  w += (size_t)HGRU3 * 4;                     //     23,040
    float* hbuf     = (float*)w;  w += 2 * HGRU * 4;                          //     15,360
    u32*   bar      = (u32*)w;    w += 256;

    cast_weights_bf16<<<2048, 256, 0, stream>>>(w_ih, w_hh, wih_bf, whh_bf,
                                                HGRU3 * HGRU);
    conv1_sig_pool<<<NFRAMES, 256, 0, stream>>>(x, conv1_w, conv1_b, conv1out);
    conv2_sig_pool<<<NFRAMES, 256, 0, stream>>>(conv1out, conv2_w, conv2_b,
                                                feats_bf);
    xproj_gemm_wmma<<<dim3(HGRU3 / 128, NFRAMES / 128), 256,
                      2 * GEMM_BUF_BYTES, stream>>>(feats_bf, wih_bf, b_ih,
                                                    xproj);
    gru_init<<<(HGRU + 255) / 256, 256, 0, stream>>>(h0, hbuf, bar);
    gru_scan<<<NWG_GRU, 256, 0, stream>>>(whh_bf, b_hh, xproj, hbuf, hp, bar,
                                          NFRAMES);
    // T=2048 (even) -> final h is in parity-0 half of hbuf
    fc_head<<<1, 64, 0, stream>>>(hbuf, fc_w, fc_b, out);
}
